// GaussianImageCrossAttention_60241211293721
// MI455X (gfx1250) — matrix operand order repaired and checked by reading, not compiled, not run on hardware
//
#include <hip/hip_runtime.h>
#include <hip/hip_fp16.h>
#include <cstdint>

typedef _Float16 h8  __attribute__((ext_vector_type(8)));
typedef _Float16 h16 __attribute__((ext_vector_type(16)));
typedef float    f8  __attribute__((ext_vector_type(8)));
typedef uint32_t u32x4 __attribute__((ext_vector_type(4)));
typedef uint32_t u32x8 __attribute__((ext_vector_type(8)));

#define NQ    6400
#define CC    256
#define NCAM  6
#define NH    8
#define NL    3
#define NP    8
#define S_TOT 14784   // 11264 + 2816 + 704
#define KK    256     // reduction dim of every GEMM
#define LDST  264     // LDS row stride in halves: 256 + 8 (TDM pad: 4 DW / 128 DW)

// ---------------------------------------------------------------------------
// Tensor Data Mover: DMA one [64 rows x 256 halves] fp16 panel (row-major,
// row stride = 256 elements) from global into LDS, inserting 16B of padding
// after every 512B row => LDS row stride 264 halves (bank-conflict-free).
// D# per CDNA5 ISA (2-group form, 2-D tile).
// ---------------------------------------------------------------------------
__device__ __forceinline__ void tdm_load_panel(uint32_t lds_off,
                                               const _Float16* gsrc) {
  uint64_t ga = (uint64_t)(uintptr_t)gsrc;
  // group0: count=1 | lds_addr | global_addr[56:0] | type=2
  u32x4 g0 = {1u,
              lds_off,
              (uint32_t)ga,
              (uint32_t)((ga >> 32) & 0x01FFFFFFu) | (2u << 30)};
  // group1: data_size=2B(1) | pad_enable | pad_interval=128DW(6) | pad_amount=4DW(3)
  uint32_t w0 = (1u << 16) | (1u << 20) | (6u << 22) | (3u << 25);
  u32x8 g1 = {w0,
              (uint32_t)(KK & 0xFFFF) << 16,   // tensor_dim0 lo16 (=256)
              64u << 16,                       // tensor_dim0 hi=0 | tensor_dim1 lo16 (=64)
              (uint32_t)KK << 16,              // tensor_dim1 hi=0 | tile_dim0 (=256)
              64u,                             // tile_dim1 (=64) | tile_dim2=0
              (uint32_t)KK,                    // tensor_dim0_stride lo32 (=256 elems)
              0u,
              0u};
  asm volatile("tensor_load_to_lds %0, %1" ::"s"(g0), "s"(g1) : "memory");
}

// ---------------------------------------------------------------------------
// WMMA GEMM with TDM-staged panels:
//   C[M,N] = A[M,256](fp16) * Bt[N,256](fp16)^T + bias[N]
// Block: 256 thr (8 waves), tile 64x64, whole-K panels resident in LDS
// (2 x 33 KB). No staging loops, no k-loop barriers: ds_load_b128 + v_wmma.
// M,N multiples of 64.
// ---------------------------------------------------------------------------
template <typename OUT>
__global__ __launch_bounds__(256) void gemm_tdm_wmma(
    const _Float16* __restrict__ A, const _Float16* __restrict__ Bt,
    const float* __restrict__ bias, OUT* __restrict__ C, int N) {
  __shared__ __align__(16) _Float16 As[64 * LDST];
  __shared__ __align__(16) _Float16 Bs[64 * LDST];
  const int t  = threadIdx.x;
  const int m0 = blockIdx.y * 64;
  const int n0 = blockIdx.x * 64;

  if (t < 32) {  // wave 0 issues both panel DMAs, waits TENSORcnt==0
    tdm_load_panel((uint32_t)(uintptr_t)As, A + (size_t)m0 * KK);
    tdm_load_panel((uint32_t)(uintptr_t)Bs, Bt + (size_t)n0 * KK);
    __builtin_amdgcn_s_wait_tensorcnt(0);
  }
  __syncthreads();

  const int wv    = t >> 5;
  const int lane  = t & 31;
  const int lm    = lane & 15;
  const int half  = lane >> 4;
  const int strip = wv & 3;   // 16-row strip
  const int nh    = wv >> 2;  // 32-col half

  f8 acc0 = {0.f, 0.f, 0.f, 0.f, 0.f, 0.f, 0.f, 0.f};
  f8 acc1 = acc0;
  const int am = strip * 16 + lm;
  const int bn0 = nh * 32 + lm;

#pragma unroll
  for (int k0 = 0; k0 < KK; k0 += 32) {
    // A frag: a[0..7]=A[m][k0+half*8+0..7], a[8..15]=A[m][k0+16+half*8+0..7]
    h8 alo = *(const h8*)(As + am * LDST + k0 + half * 8);
    h8 ahi = *(const h8*)(As + am * LDST + k0 + 16 + half * 8);
    h16 av;
#pragma unroll
    for (int i = 0; i < 8; ++i) { av[i] = alo[i]; av[i + 8] = ahi[i]; }
#pragma unroll
    for (int ct = 0; ct < 2; ++ct) {
      // B frag: b[0..7]=Bt[n][k0+half*8+0..7], b[8..15]=Bt[n][k0+16+half*8+0..7]
      const _Float16* bp = Bs + (bn0 + ct * 16) * LDST + k0 + half * 8;
      h8 blo = *(const h8*)(bp);
      h8 bhi = *(const h8*)(bp + 16);
      h16 bv;
#pragma unroll
      for (int i = 0; i < 8; ++i) { bv[i] = blo[i]; bv[i + 8] = bhi[i]; }
      if (ct == 0)
        acc0 = __builtin_amdgcn_wmma_f32_16x16x32_f16(false, av, false, bv,
                                                      (short)0, acc0, false, false);
      else
        acc1 = __builtin_amdgcn_wmma_f32_16x16x32_f16(false, av, false, bv,
                                                      (short)0, acc1, false, false);
    }
  }
#pragma unroll
  for (int ct = 0; ct < 2; ++ct) {
    const f8& acc = ct ? acc1 : acc0;
    int gn = n0 + nh * 32 + ct * 16 + lm;
    float bb = bias[gn];
#pragma unroll
    for (int r = 0; r < 8; ++r) {
      int gm = m0 + strip * 16 + half * 8 + r;
      C[(size_t)gm * N + gn] = (OUT)(acc[r] + bb);
    }
  }
}

// --------------------------- prep kernels ----------------------------------
// feature f32 -> fp16 (contiguous)
__global__ void fcvt_kernel(const float* __restrict__ src,
                            _Float16* __restrict__ dst, int count) {
  int i = blockIdx.x * 256 + threadIdx.x;
  if (i < count) dst[i] = (_Float16)src[i];
}

// W[K=256][N] f32 -> Wt[N][256] fp16 (transpose + convert; tiny matrices)
__global__ void wtrans_kernel(const float* __restrict__ W,
                              _Float16* __restrict__ Wt, int N) {
  int idx = blockIdx.x * 256 + threadIdx.x;
  if (idx >= N * KK) return;
  int k = idx & 255, n = idx >> 8;
  Wt[idx] = (_Float16)W[(size_t)k * N + n];
}

// feats [cam][c][s] f32 -> Xv[(cam*S_TOT + s)][c] fp16, fusing cam/level
// embeddings.  LDS 32x32 tile transpose; both sides coalesced.
__global__ __launch_bounds__(256) void xv_kernel(
    const float* __restrict__ f0, const float* __restrict__ f1,
    const float* __restrict__ f2, const float* __restrict__ cemb,
    const float* __restrict__ lemb, _Float16* __restrict__ Xv) {
  __shared__ _Float16 T[32 * 34];
  const int cam = blockIdx.z;
  const int c0 = blockIdx.x * 32;
  int yb = blockIdx.y;
  int lvl, s0, lbase, HW;
  const float* fp;
  if (yb < 352)      { lvl = 0; s0 = yb * 32;         lbase = 0;     HW = 11264; fp = f0; }
  else if (yb < 440) { lvl = 1; s0 = (yb - 352) * 32; lbase = 11264; HW = 2816;  fp = f1; }
  else               { lvl = 2; s0 = (yb - 440) * 32; lbase = 14080; HW = 704;   fp = f2; }
  const int tx = threadIdx.x & 31, ty = threadIdx.x >> 5;
#pragma unroll
  for (int i = 0; i < 4; ++i) {
    int cl = ty + i * 8;
    int c = c0 + cl;
    float e = cemb[cam * CC + c] + lemb[lvl * CC + c];
    float v = fp[((size_t)cam * CC + c) * HW + s0 + tx];
    T[cl * 34 + tx] = (_Float16)(v + e);
  }
  __syncthreads();
#pragma unroll
  for (int i = 0; i < 4; ++i) {
    int sl = ty + i * 8;
    size_t gr = (size_t)cam * S_TOT + lbase + s0 + sl;
    Xv[gr * CC + c0 + tx] = T[tx * 34 + sl];
  }
}

// ------------------------- geometry / softmax ------------------------------
__global__ void geo_kernel(const float* __restrict__ means,
                           const float* __restrict__ cam2ego,
                           const float* __restrict__ intrins,
                           const float* __restrict__ post_rots,
                           const float* __restrict__ post_trans,
                           const int* __restrict__ img_h,
                           const int* __restrict__ img_w,
                           float* __restrict__ coor, float* __restrict__ maskf) {
  int n = blockIdx.x * 256 + threadIdx.x;
  if (n >= NQ) return;
  int cam = blockIdx.y;
  const float* M4 = cam2ego + cam * 16;
  const float* K9 = intrins + cam * 9;
  const float* P9 = post_rots + cam * 9;
  const float* pt = post_trans + cam * 3;
  float mx = means[n * 3 + 0], my = means[n * 3 + 1], mz = means[n * 3 + 2];
  float px = mx - M4[3], py = my - M4[7], pz = mz - M4[11];
  float cx = M4[0] * px + M4[4] * py + M4[8] * pz;
  float cy = M4[1] * px + M4[5] * py + M4[9] * pz;
  float cz = M4[2] * px + M4[6] * py + M4[10] * pz;
  float ix = K9[0] * cx + K9[1] * cy + K9[2] * cz;
  float iy = K9[3] * cx + K9[4] * cy + K9[5] * cz;
  float iz = K9[6] * cx + K9[7] * cy + K9[8] * cz;
  float inv = 1.0f / (iz + 1e-4f);
  float u = ix * inv, v = iy * inv;
  float qx = P9[0] * u + P9[1] * v + P9[2] * iz + pt[0];
  float qy = P9[3] * u + P9[4] * v + P9[5] * iz + pt[1];
  float qz = P9[6] * u + P9[7] * v + P9[8] * iz + pt[2];
  float cxo = qx / (float)(*img_w);
  float cyo = qy / (float)(*img_h);
  bool ok = (qz > 0.01f) && (cxo > 0.f) && (cxo < 1.f) && (cyo > 0.f) && (cyo < 1.f);
  size_t b = (size_t)cam * NQ + n;
  coor[b * 2 + 0] = cxo;
  coor[b * 2 + 1] = cyo;
  maskf[b] = ok ? 1.f : 0.f;
}

__global__ void softmax_kernel(const float* __restrict__ raw,
                               float* __restrict__ attn) {
  int g = blockIdx.x * 256 + threadIdx.x;
  if (g >= NQ * NH) return;
  size_t base = (size_t)g * (NL * NP);
  float m = -1e30f;
  for (int i = 0; i < NL * NP; ++i) m = fmaxf(m, raw[base + i]);
  float s = 0.f;
  for (int i = 0; i < NL * NP; ++i) {
    float e = __expf(raw[base + i] - m);
    attn[base + i] = e;
    s += e;
  }
  float r = 1.0f / s;
  for (int i = 0; i < NL * NP; ++i) attn[base + i] *= r;
}

// ---------------------------------------------------------------------------
// Deformable sampler. Block=(query,cam); wave=head; lane=channel. 32 lanes
// gather 32 contiguous fp16 channels (64B) per tap; value fp16 (45MB) is
// L2-resident (192MB L2).
// ---------------------------------------------------------------------------
__global__ __launch_bounds__(256) void sample_kernel(
    const float* __restrict__ coor, const float* __restrict__ offs,
    const float* __restrict__ attn, const _Float16* __restrict__ value,
    _Float16* __restrict__ out_cams) {
  const int n = blockIdx.x, cam = blockIdx.y;
  const int h = threadIdx.x >> 5, d = threadIdx.x & 31;
  const int lw[3] = {176, 88, 44}, lh[3] = {64, 32, 16}, lb[3] = {0, 11264, 14080};
  float cx = coor[((size_t)cam * NQ + n) * 2 + 0];
  float cy = coor[((size_t)cam * NQ + n) * 2 + 1];
  const float* op = offs + (size_t)n * (NH * NL * NP * 2) + h * (NL * NP * 2);
  const float* ap = attn + (size_t)n * (NH * NL * NP) + h * (NL * NP);
  const size_t vbase = (size_t)cam * S_TOT;
  const int ch = (h << 5) + d;
  float acc = 0.f;
#pragma unroll
  for (int lvl = 0; lvl < NL; ++lvl) {
    const int w = lw[lvl], hh = lh[lvl], bb = lb[lvl];
    const float fw = (float)w, fh = (float)hh;
#pragma unroll
    for (int p = 0; p < NP; ++p) {
      float ox = op[(lvl * NP + p) * 2 + 0];
      float oy = op[(lvl * NP + p) * 2 + 1];
      float a = ap[lvl * NP + p];
      float x = (cx + ox / fw) * fw - 0.5f;
      float y = (cy + oy / fh) * fh - 0.5f;
      float x0 = floorf(x), y0 = floorf(y);
      float tx = x - x0, ty = y - y0;
      float wts[4] = {(1.f - tx) * (1.f - ty), tx * (1.f - ty),
                      (1.f - tx) * ty, tx * ty};
      float xs[4] = {x0, x0 + 1.f, x0, x0 + 1.f};
      float ys[4] = {y0, y0, y0 + 1.f, y0 + 1.f};
#pragma unroll
      for (int tp = 0; tp < 4; ++tp) {
        float xi = xs[tp], yi = ys[tp];
        bool valid = (xi >= 0.f) && (xi <= fw - 1.f) && (yi >= 0.f) && (yi <= fh - 1.f);
        int xic = (int)fminf(fmaxf(xi, 0.f), fw - 1.f);
        int yic = (int)fminf(fmaxf(yi, 0.f), fh - 1.f);
        size_t r = vbase + bb + (size_t)yic * w + xic;
        float g = (float)value[r * CC + ch];
        acc += g * (wts[tp] * a * (valid ? 1.f : 0.f));
      }
    }
  }
  out_cams[((size_t)cam * NQ + n) * CC + ch] = (_Float16)acc;
}

// Masked camera average (fixed 6-term order -> deterministic); fp16 out for GEMM.
__global__ void reduce_kernel(const _Float16* __restrict__ out_cams,
                              const float* __restrict__ maskf,
                              _Float16* __restrict__ slots) {
  int n = blockIdx.x, c = threadIdx.x;
  float num = 0.f, den = 0.f;
#pragma unroll
  for (int cam = 0; cam < NCAM; ++cam) {
    float m = maskf[(size_t)cam * NQ + n];
    den += m;
    num += m * (float)out_cams[((size_t)cam * NQ + n) * CC + c];
  }
  slots[(size_t)n * CC + c] = (_Float16)(num / fmaxf(den, 1.f));
}

// Residual + LayerNorm over C=256.
__global__ __launch_bounds__(256) void ln_kernel(
    const float* __restrict__ y, const float* __restrict__ feat,
    const float* __restrict__ g, const float* __restrict__ b,
    float* __restrict__ out) {
  __shared__ float sb[256];
  int n = blockIdx.x, c = threadIdx.x;
  float x = y[(size_t)n * CC + c] + feat[(size_t)n * CC + c];
  sb[c] = x;
  __syncthreads();
  for (int s = 128; s > 0; s >>= 1) {
    if (c < s) sb[c] += sb[c + s];
    __syncthreads();
  }
  float mu = sb[0] * (1.f / 256.f);
  __syncthreads();
  float dx = x - mu;
  sb[c] = dx * dx;
  __syncthreads();
  for (int s = 128; s > 0; s >>= 1) {
    if (c < s) sb[c] += sb[c + s];
    __syncthreads();
  }
  float var = sb[0] * (1.f / 256.f);
  out[(size_t)n * CC + c] = dx * rsqrtf(var + 1e-5f) * g[c] + b[c];
}

// ---------------------------------------------------------------------------
extern "C" void kernel_launch(void* const* d_in, const int* in_sizes, int n_in,
                              void* d_out, int out_size, void* d_ws, size_t ws_size,
                              hipStream_t stream) {
  const float* means      = (const float*)d_in[0];
  const float* feature    = (const float*)d_in[1];
  const float* feat0      = (const float*)d_in[2];
  const float* feat1      = (const float*)d_in[3];
  const float* feat2      = (const float*)d_in[4];
  const float* cam2ego    = (const float*)d_in[5];
  const float* intrins    = (const float*)d_in[6];
  const float* post_rots  = (const float*)d_in[7];
  const float* post_trans = (const float*)d_in[8];
  const float* W_off      = (const float*)d_in[9];
  const float* b_off      = (const float*)d_in[10];
  const float* W_attn     = (const float*)d_in[11];
  const float* b_attn     = (const float*)d_in[12];
  const float* W_val      = (const float*)d_in[13];
  const float* b_val      = (const float*)d_in[14];
  const float* W_out      = (const float*)d_in[15];
  const float* b_out      = (const float*)d_in[16];
  const float* cemb       = (const float*)d_in[17];
  const float* lemb       = (const float*)d_in[18];
  const float* ln_g       = (const float*)d_in[19];
  const float* ln_b       = (const float*)d_in[20];
  const int*   img_h      = (const int*)d_in[21];
  const int*   img_w      = (const int*)d_in[22];
  float* out = (float*)d_out;

  char* wp = (char*)d_ws;
  auto alloc = [&](size_t bytes) {
    char* p = wp;
    wp += (bytes + 255) & ~(size_t)255;
    return p;
  };
  float*    coor     = (float*)alloc((size_t)NCAM * NQ * 2 * 4);
  float*    maskf    = (float*)alloc((size_t)NCAM * NQ * 4);
  _Float16* Xq       = (_Float16*)alloc((size_t)NQ * CC * 2);              // 3.3 MB
  _Float16* Wt_off   = (_Float16*)alloc((size_t)384 * KK * 2);
  _Float16* Wt_attn  = (_Float16*)alloc((size_t)192 * KK * 2);
  _Float16* Wt_val   = (_Float16*)alloc((size_t)256 * KK * 2);
  _Float16* Wt_out   = (_Float16*)alloc((size_t)256 * KK * 2);
  float*    offs     = (float*)alloc((size_t)NQ * NH * NL * NP * 2 * 4);   // 9.8 MB
  float*    attn_raw = (float*)alloc((size_t)NQ * NH * NL * NP * 4);       // 4.9 MB
  float*    attn     = (float*)alloc((size_t)NQ * NH * NL * NP * 4);       // 4.9 MB
  _Float16* Xv       = (_Float16*)alloc((size_t)NCAM * S_TOT * CC * 2);    // 45.4 MB
  _Float16* value    = (_Float16*)alloc((size_t)NCAM * S_TOT * CC * 2);    // 45.4 MB
  _Float16* out_cams = (_Float16*)alloc((size_t)NCAM * NQ * CC * 2);       // 19.7 MB
  _Float16* slots    = (_Float16*)alloc((size_t)NQ * CC * 2);              // 3.3 MB
  float*    yv       = (float*)alloc((size_t)NQ * CC * 4);                 // 6.6 MB

  geo_kernel<<<dim3((NQ + 255) / 256, NCAM), 256, 0, stream>>>(
      means, cam2ego, intrins, post_rots, post_trans, img_h, img_w, coor, maskf);

  // fp16 operand prep
  fcvt_kernel<<<dim3((NQ * CC) / 256), 256, 0, stream>>>(feature, Xq, NQ * CC);
  wtrans_kernel<<<dim3(384 * KK / 256), 256, 0, stream>>>(W_off, Wt_off, 384);
  wtrans_kernel<<<dim3(192 * KK / 256), 256, 0, stream>>>(W_attn, Wt_attn, 192);
  wtrans_kernel<<<dim3(256 * KK / 256), 256, 0, stream>>>(W_val, Wt_val, 256);
  wtrans_kernel<<<dim3(256 * KK / 256), 256, 0, stream>>>(W_out, Wt_out, 256);
  xv_kernel<<<dim3(CC / 32, 462, NCAM), 256, 0, stream>>>(feat0, feat1, feat2,
                                                          cemb, lemb, Xv);

  // offsets / attention logits (TDM + WMMA)
  gemm_tdm_wmma<float><<<dim3(384 / 64, NQ / 64), 256, 0, stream>>>(
      Xq, Wt_off, b_off, offs, 384);
  gemm_tdm_wmma<float><<<dim3(192 / 64, NQ / 64), 256, 0, stream>>>(
      Xq, Wt_attn, b_attn, attn_raw, 192);
  softmax_kernel<<<dim3((NQ * NH) / 256), 256, 0, stream>>>(attn_raw, attn);

  // value projection (TDM + WMMA), fp16 output, L2-resident
  gemm_tdm_wmma<_Float16><<<dim3(CC / 64, (NCAM * S_TOT) / 64), 256, 0, stream>>>(
      Xv, Wt_val, b_val, value, CC);

  sample_kernel<<<dim3(NQ, NCAM), 256, 0, stream>>>(coor, offs, attn, value,
                                                    out_cams);
  reduce_kernel<<<dim3(NQ), 256, 0, stream>>>(out_cams, maskf, slots);

  // output projection (TDM + WMMA)
  gemm_tdm_wmma<float><<<dim3(CC / 64, NQ / 64), 256, 0, stream>>>(
      slots, Wt_out, b_out, yv, CC);

  ln_kernel<<<dim3(NQ), 256, 0, stream>>>(yv, feature, ln_g, ln_b, out);
}
